// PAM_45646912422434
// MI455X (gfx1250) — matrix operand order repaired and checked by reading, not compiled, not run on hardware
//
#include <hip/hip_runtime.h>
#include <hip/hip_bf16.h>

typedef __attribute__((ext_vector_type(16))) _Float16 v16h;
typedef __attribute__((ext_vector_type(8)))  _Float16 v8h;
typedef __attribute__((ext_vector_type(4)))  _Float16 h4;
typedef __attribute__((ext_vector_type(8)))  float    v8f;

#define N_TOK 4096
#define N_CH  64
#define N_B   4

// ---------------------------------------------------------------------------
// Kernel 1: joint[b][j] = mean over channels of x[b,c,j]
// ---------------------------------------------------------------------------
__global__ __launch_bounds__(256)
void k_joint(const float* __restrict__ x, float* __restrict__ joint) {
  const int j = blockIdx.x * 256 + threadIdx.x;
  const int b = blockIdx.y;
  const float* xp = x + (size_t)b * N_CH * N_TOK + j;
  float s = 0.f;
#pragma unroll 8
  for (int c = 0; c < N_CH; ++c) s += xp[(size_t)c * N_TOK];
  joint[b * N_TOK + j] = s * (1.0f / N_CH);
}

// ---------------------------------------------------------------------------
// Kernel 2: per-batch max/min of joint (for stable softmax row-max)
// ---------------------------------------------------------------------------
__global__ __launch_bounds__(256)
void k_minmax(const float* __restrict__ joint, float* __restrict__ gmax,
              float* __restrict__ gmin) {
  __shared__ float smax[256];
  __shared__ float smin[256];
  const int b = blockIdx.x;
  const int t = threadIdx.x;
  float mx = -3.402823466e38f, mn = 3.402823466e38f;
  for (int j = t; j < N_TOK; j += 256) {
    float v = joint[b * N_TOK + j];
    mx = fmaxf(mx, v);
    mn = fminf(mn, v);
  }
  smax[t] = mx; smin[t] = mn;
  __syncthreads();
  for (int s = 128; s > 0; s >>= 1) {
    if (t < s) {
      smax[t] = fmaxf(smax[t], smax[t + s]);
      smin[t] = fminf(smin[t], smin[t + s]);
    }
    __syncthreads();
  }
  if (t == 0) { gmax[b] = smax[0]; gmin[b] = smin[0]; }
}

// ---------------------------------------------------------------------------
// Kernel 3: row max m_j = max_k(q_j*q_k) and Z_j = sum_k exp(q_j*q_k - m_j)
// ---------------------------------------------------------------------------
__global__ __launch_bounds__(256)
void k_z(const float* __restrict__ joint, const float* __restrict__ gmax,
         const float* __restrict__ gmin, float* __restrict__ mj_out,
         float* __restrict__ invz_out) {
  __shared__ float sj[N_TOK];
  const int b = blockIdx.y;
  for (int i = threadIdx.x; i < N_TOK / 4; i += 256)
    ((float4*)sj)[i] = ((const float4*)(joint + (size_t)b * N_TOK))[i];
  __syncthreads();

  const int j = blockIdx.x * 256 + threadIdx.x;
  const float qj = sj[j];
  // max_k q_j*q_k is q_j*gmax when q_j>=0, else q_j*gmin (rank-1 scores)
  const float mj = (qj >= 0.f) ? qj * gmax[b] : qj * gmin[b];
  float z = 0.f;
  for (int k = 0; k < N_TOK; k += 4) {
    float4 q = *(const float4*)&sj[k];
    z += __expf(__builtin_fmaf(qj, q.x, -mj));
    z += __expf(__builtin_fmaf(qj, q.y, -mj));
    z += __expf(__builtin_fmaf(qj, q.z, -mj));
    z += __expf(__builtin_fmaf(qj, q.w, -mj));
  }
  mj_out[b * N_TOK + j]   = mj;
  invz_out[b * N_TOK + j] = 1.f / z;
}

// ---------------------------------------------------------------------------
// Kernel 4: convert x (f32) -> f16 copy for WMMA A-matrix feeds
// ---------------------------------------------------------------------------
__global__ __launch_bounds__(256)
void k_cvt(const float* __restrict__ x, _Float16* __restrict__ vh) {
  const size_t i = (size_t)blockIdx.x * 256 + threadIdx.x; // in float4 units
  float4 v = ((const float4*)x)[i];
  h4 o = { (_Float16)v.x, (_Float16)v.y, (_Float16)v.z, (_Float16)v.w };
  ((h4*)vh)[i] = o;
}

// ---------------------------------------------------------------------------
// Kernel 5: fused softmax-weight regeneration + WMMA GEMM + residual.
// out[b,c,j] = (sum_k V[b,c,k] * exp(q_j*q_k - m_j)) * invZ_j + x[b,c,j]
// Block: 128 thr (4 waves). Wave w handles j-slice [blk*64 + w*16, +16),
// all 4 c-tiles (c = 0..63). K loop unrolled x2 (64/iter) with ping-pong
// A-fragment buffers so global loads overlap exp-VALU + WMMA with no
// register-rotation moves.
// ---------------------------------------------------------------------------
__global__ __launch_bounds__(128)
void k_gemm(const float* __restrict__ x, const _Float16* __restrict__ vh,
            const float* __restrict__ joint, const float* __restrict__ mjv,
            const float* __restrict__ invzv, float* __restrict__ out) {
  __shared__ float sj[N_TOK];
  const int b = blockIdx.y;
  for (int i = threadIdx.x; i < N_TOK / 4; i += 128)
    ((float4*)sj)[i] = ((const float4*)(joint + (size_t)b * N_TOK))[i];
  __syncthreads();

  const int lane = threadIdx.x & 31;
  const int wave = threadIdx.x >> 5;
  const int kh   = lane >> 4;   // which K-half this lane feeds (A/B layout)
  const int nn   = lane & 15;   // row (A) / column (B,D) owned by this lane
  const int j    = blockIdx.x * 64 + wave * 16 + nn;

  const float qj = sj[j];
  const float mj = mjv[(size_t)b * N_TOK + j];
  const float iz = invzv[(size_t)b * N_TOK + j];

  union BF { v16h v; _Float16 e[16]; };
  union AF { v16h v; v8h h[2]; };

  // Per-lane base pointer into the f16 V panel for this batch.
  const _Float16* apb =
      vh + (size_t)b * N_CH * N_TOK + (size_t)nn * N_TOK + kh * 8;

// Load the four c-tile A fragments for K-offset (koff) into d0..d3.
#define LOAD_A(d0, d1, d2, d3, koff)                                     \
  do {                                                                   \
    const _Float16* _ap = apb + (koff);                                  \
    d0.h[0] = *(const v8h*)(_ap);                                        \
    d0.h[1] = *(const v8h*)(_ap + 16);                                   \
    d1.h[0] = *(const v8h*)(_ap + (size_t)16 * N_TOK);                   \
    d1.h[1] = *(const v8h*)(_ap + (size_t)16 * N_TOK + 16);              \
    d2.h[0] = *(const v8h*)(_ap + (size_t)32 * N_TOK);                   \
    d2.h[1] = *(const v8h*)(_ap + (size_t)32 * N_TOK + 16);              \
    d3.h[0] = *(const v8h*)(_ap + (size_t)48 * N_TOK);                   \
    d3.h[1] = *(const v8h*)(_ap + (size_t)48 * N_TOK + 16);              \
  } while (0)

// Build the 32x16 exp-weight B fragment for K-offset (koff) into bf.
#define BUILD_B(bf, koff)                                                \
  do {                                                                   \
    float _qk[16];                                                       \
    const float4* _qp = (const float4*)&sj[(koff) + kh * 8];             \
    *(float4*)&_qk[0]  = _qp[0];                                         \
    *(float4*)&_qk[4]  = _qp[1];                                         \
    *(float4*)&_qk[8]  = _qp[4];                                         \
    *(float4*)&_qk[12] = _qp[5];                                         \
    _Pragma("unroll")                                                    \
    for (int _e = 0; _e < 16; ++_e)                                      \
      bf.e[_e] = (_Float16)__expf(__builtin_fmaf(qj, _qk[_e], -mj));     \
  } while (0)

#define WMMA4(a0, a1, a2, a3, bf)                                        \
  do {                                                                   \
    acc0 = __builtin_amdgcn_wmma_f32_16x16x32_f16(                       \
        false, a0.v, false, bf.v, (short)0, acc0, false, false);         \
    acc1 = __builtin_amdgcn_wmma_f32_16x16x32_f16(                       \
        false, a1.v, false, bf.v, (short)0, acc1, false, false);         \
    acc2 = __builtin_amdgcn_wmma_f32_16x16x32_f16(                       \
        false, a2.v, false, bf.v, (short)0, acc2, false, false);         \
    acc3 = __builtin_amdgcn_wmma_f32_16x16x32_f16(                       \
        false, a3.v, false, bf.v, (short)0, acc3, false, false);         \
  } while (0)

  v8f acc0 = {}, acc1 = {}, acc2 = {}, acc3 = {};

  AF pa0, pa1, pa2, pa3;  // ping buffer (even K-steps)
  LOAD_A(pa0, pa1, pa2, pa3, 0);

  for (int k0 = 0; k0 < N_TOK; k0 += 64) {
    // Pong buffer loads (k0+32) issue before even-step compute.
    AF pb0, pb1, pb2, pb3;
    LOAD_A(pb0, pb1, pb2, pb3, k0 + 32);

    BF bf0;
    BUILD_B(bf0, k0);
    WMMA4(pa0, pa1, pa2, pa3, bf0);

    // Ping buffer loads for next iteration (wrap-safe) overlap odd-step compute.
    LOAD_A(pa0, pa1, pa2, pa3, (k0 + 64) & (N_TOK - 1));

    BF bf1;
    BUILD_B(bf1, k0 + 32);
    WMMA4(pb0, pb1, pb2, pb3, bf1);
  }

#undef LOAD_A
#undef BUILD_B
#undef WMMA4

  // ---- Normalize, add residual, store. D layout: elem r -> row kh*8+r, col nn.
  const size_t base = (size_t)b * N_CH * N_TOK + j;
#pragma unroll
  for (int r = 0; r < 8; ++r) {
    const size_t i0 = base + (size_t)(kh * 8 + r) * N_TOK;
    out[i0]                        = acc0[r] * iz + x[i0];
    out[i0 + (size_t)16 * N_TOK]   = acc1[r] * iz + x[i0 + (size_t)16 * N_TOK];
    out[i0 + (size_t)32 * N_TOK]   = acc2[r] * iz + x[i0 + (size_t)32 * N_TOK];
    out[i0 + (size_t)48 * N_TOK]   = acc3[r] * iz + x[i0 + (size_t)48 * N_TOK];
  }
}

// ---------------------------------------------------------------------------
extern "C" void kernel_launch(void* const* d_in, const int* in_sizes, int n_in,
                              void* d_out, int out_size, void* d_ws, size_t ws_size,
                              hipStream_t stream) {
  const float* x = (const float*)d_in[0];
  float* out = (float*)d_out;

  char* ws = (char*)d_ws;
  float* joint = (float*)ws;                       // 4*4096 f32  (64 KB)
  float* mj    = joint + N_B * N_TOK;              // 4*4096 f32
  float* invz  = mj    + N_B * N_TOK;              // 4*4096 f32
  float* gmax  = invz  + N_B * N_TOK;              // 4 f32
  float* gmin  = gmax + 4;                         // 4 f32
  _Float16* vh = (_Float16*)(ws + 3 * (size_t)N_B * N_TOK * 4 + 64); // 2 MB f16

  k_joint <<<dim3(N_TOK / 256, N_B), 256, 0, stream>>>(x, joint);
  k_minmax<<<N_B, 256, 0, stream>>>(joint, gmax, gmin);
  k_z     <<<dim3(N_TOK / 256, N_B), 256, 0, stream>>>(joint, gmax, gmin, mj, invz);
  k_cvt   <<<(N_B * N_CH * N_TOK) / 4 / 256, 256, 0, stream>>>(x, vh);
  k_gemm  <<<dim3(N_TOK / 64, N_B), 128, 0, stream>>>(x, vh, joint, mj, invz, out);
}
